// CLAM_33234456936750
// MI455X (gfx1250) — matrix operand order repaired and checked by reading, not compiled, not run on hardware
//
#include <hip/hip_runtime.h>
#include <cstdint>
#include <cstddef>

// ---------------- problem constants ----------------
#define NN 50000
#define LD 1024
#define HD 512
#define DD 256
#define KS 8

typedef __bf16 bf16_t;
typedef __attribute__((ext_vector_type(16))) __bf16 v16bf;
typedef __attribute__((ext_vector_type(8)))  __bf16 v8bf;
typedef __attribute__((ext_vector_type(8)))  float  v8f;

static __device__ __forceinline__ v16bf cat8(v8bf lo, v8bf hi) {
  return __builtin_shufflevector(lo, hi, 0,1,2,3,4,5,6,7,8,9,10,11,12,13,14,15);
}

static __device__ __forceinline__ v8f wmma_bf16(v16bf a, v16bf b, v8f c) {
  return __builtin_amdgcn_wmma_f32_16x16x32_bf16(false, a, false, b, (short)0, c, false, false);
}

// gfx1250 async global->LDS copy, 16B per lane; tracked by ASYNCcnt.
// ldsoff = LDS byte address (low 32 bits of flat address of a __shared__ object),
// gp     = 64-bit global address.
static __device__ __forceinline__ void async_copy_b128(uint32_t ldsoff, const void* gp) {
  asm volatile("global_load_async_to_lds_b128 %0, %1, off"
               :: "v"(ldsoff), "v"(gp) : "memory");
}
static __device__ __forceinline__ void wait_asynccnt0() {
  asm volatile("s_wait_asynccnt 0x0" ::: "memory");
}

// ---------------- kernel 0: weight prep (f32 -> bf16, transposed) ------------
// WfcT[h][l] = Wfc[l][h]; WaT[d][h] = Wa[h][d]; WbT[d][h] = Wb[h][d]
__global__ __launch_bounds__(256) void clam_prep(const float* __restrict__ Wfc,
                                                 const float* __restrict__ Wa,
                                                 const float* __restrict__ Wb,
                                                 bf16_t* __restrict__ WfcT,
                                                 bf16_t* __restrict__ WaT,
                                                 bf16_t* __restrict__ WbT) {
  int t = blockIdx.x * 256 + threadIdx.x;
  if (t < LD * HD) {            // 524288
    int hcol = t / LD, l = t % LD;
    WfcT[t] = (bf16_t)Wfc[(size_t)l * HD + hcol];
  }
  if (t < HD * DD) {            // 131072
    int d = t / HD, hh = t % HD;
    WaT[t] = (bf16_t)Wa[(size_t)hh * DD + d];
    WbT[t] = (bf16_t)Wb[(size_t)hh * DD + d];
  }
}

// ---------------- kernel 1: x = relu(h @ Wfc + bfc), store bf16 --------------
// Block: 256 thr = 8 waves; tile 128(M) x 64(N); wave = 32x32 (2x2 WMMA tiles)
// B tile (64 cols x 32 K bf16 = 4KB) staged per K-step via async global->LDS.
__global__ __launch_bounds__(256) void clam_gemm1(const float* __restrict__ h,
                                                  const bf16_t* __restrict__ WfcT,
                                                  const float* __restrict__ bfc,
                                                  bf16_t* __restrict__ x) {
  __shared__ __align__(16) bf16_t bsm[64 * 32];   // [col][k]

  const int tid  = threadIdx.x;
  const int lane = tid & 31;
  const int wave = tid >> 5;
  const int half = lane >> 4;      // K-half select for A/B fragments
  const int l16  = lane & 15;
  const int mtile = blockIdx.x >> 3;     // 391 tiles of 128 rows
  const int ntile = blockIdx.x & 7;      // 8 tiles of 64 cols
  const int m0    = mtile * 128 + (wave & 3) * 32;
  const int nbase = ntile * 64;
  const int n0loc = (wave >> 2) * 32;    // wave's 32 cols within block tile

  // async-staging assignment: thread -> (column, 16B chunk) of the B tile
  const int scol   = tid >> 2;     // 0..63
  const int schunk = tid & 3;      // 0..3  (4 x 16B = 32 K-elements row)
  const bf16_t* bsrc = WfcT + (size_t)(nbase + scol) * LD + schunk * 8;
  const uint32_t ldsdst = (uint32_t)(uintptr_t)(&bsm[scol * 32 + schunk * 8]);

  // loop-invariant A row pointers (clamped; garbage rows never stored)
  const float* arow[2];
#pragma unroll
  for (int mi = 0; mi < 2; ++mi) {
    int r = m0 + mi * 16 + l16; if (r >= NN) r = NN - 1;
    arow[mi] = h + (size_t)r * LD + half * 8;
  }

  v8f acc[2][2] = {};

  for (int kb = 0; kb < LD; kb += 32) {
    // ---- kick off async B-tile fill (16B per thread, ASYNCcnt-tracked) ----
    async_copy_b128(ldsdst, (const void*)(bsrc + kb));

    // ---- A fragments: f32 loads overlap the async fill; convert to bf16 ----
    v16bf afrag[2];
#pragma unroll
    for (int mi = 0; mi < 2; ++mi) {
      const float* p = arow[mi] + kb;
      float4 f0 = *(const float4*)(p);
      float4 f1 = *(const float4*)(p + 4);
      float4 f2 = *(const float4*)(p + 16);
      float4 f3 = *(const float4*)(p + 20);
      __builtin_prefetch(p + 32, 0, 1);   // gfx1250 global_prefetch_b8
      v16bf a;
      a[0]=(bf16_t)f0.x;  a[1]=(bf16_t)f0.y;  a[2]=(bf16_t)f0.z;  a[3]=(bf16_t)f0.w;
      a[4]=(bf16_t)f1.x;  a[5]=(bf16_t)f1.y;  a[6]=(bf16_t)f1.z;  a[7]=(bf16_t)f1.w;
      a[8]=(bf16_t)f2.x;  a[9]=(bf16_t)f2.y;  a[10]=(bf16_t)f2.z; a[11]=(bf16_t)f2.w;
      a[12]=(bf16_t)f3.x; a[13]=(bf16_t)f3.y; a[14]=(bf16_t)f3.z; a[15]=(bf16_t)f3.w;
      afrag[mi] = a;
    }

    // ---- publish B tile: own async done, then workgroup barrier ----
    wait_asynccnt0();
    __syncthreads();

    // ---- B fragments from LDS (ds_load_b128 x2 per fragment) ----
    v16bf bfrag[2];
#pragma unroll
    for (int ni = 0; ni < 2; ++ni) {
      const bf16_t* p = &bsm[(n0loc + ni * 16 + l16) * 32 + half * 8];
      bfrag[ni] = cat8(*(const v8bf*)(p), *(const v8bf*)(p + 16));
    }

#pragma unroll
    for (int mi = 0; mi < 2; ++mi)
#pragma unroll
      for (int ni = 0; ni < 2; ++ni)
        acc[mi][ni] = wmma_bf16(afrag[mi], bfrag[ni], acc[mi][ni]);

    __syncthreads();   // all waves done reading before next fill overwrites
  }

  // ---- epilogue: bias + relu, store bf16; fast path for full tiles ----
  const bool full = (m0 + 32 <= NN);
#pragma unroll
  for (int mi = 0; mi < 2; ++mi)
#pragma unroll
    for (int ni = 0; ni < 2; ++ni) {
      int c = nbase + n0loc + ni * 16 + l16;
      float bias = bfc[c];
      if (full) {
#pragma unroll
        for (int e = 0; e < 8; ++e) {
          int r = m0 + mi * 16 + e + half * 8;
          float v = acc[mi][ni][e] + bias;
          x[(size_t)r * HD + c] = (bf16_t)(v > 0.f ? v : 0.f);
        }
      } else {
#pragma unroll
        for (int e = 0; e < 8; ++e) {
          int r = m0 + mi * 16 + e + half * 8;
          if (r < NN) {
            float v = acc[mi][ni][e] + bias;
            x[(size_t)r * HD + c] = (bf16_t)(v > 0.f ? v : 0.f);
          }
        }
      }
    }
}

// ---------------- kernel 2: raw scores A_nb[n] = (tanh(xWa+ba)*sig(xWb+bb)).Wc
// Block: 256 thr = 8 waves; 64 rows x full D=256 (no global atomics)
__global__ __launch_bounds__(256) void clam_gemm2(const bf16_t* __restrict__ x,
                                                  const bf16_t* __restrict__ WaT,
                                                  const bf16_t* __restrict__ WbT,
                                                  const float* __restrict__ ba,
                                                  const float* __restrict__ bb,
                                                  const float* __restrict__ Wc,
                                                  float* __restrict__ Anb) {
  const int lane = threadIdx.x & 31;
  const int wave = threadIdx.x >> 5;
  const int half = lane >> 4;
  const int l16  = lane & 15;
  const int m0 = blockIdx.x * 64 + (wave & 3) * 16;   // 16 rows / wave
  const int c0 = (wave >> 2) * 128;                   // column half, 8 tiles

  int r = m0 + l16; if (r >= NN) r = NN - 1;
  const bf16_t* pa0 = x + (size_t)r * HD + half * 8;  // loop-invariant base

  v8f accA[8] = {}, accB[8] = {};

  for (int kb = 0; kb < HD; kb += 32) {
    const bf16_t* pa = pa0 + kb;
    v16bf afrag = cat8(*(const v8bf*)(pa), *(const v8bf*)(pa + 16));
#pragma unroll
    for (int ni = 0; ni < 8; ++ni) {
      size_t boff = (size_t)(c0 + ni * 16 + l16) * HD + kb + half * 8;
      v16bf bA = cat8(*(const v8bf*)(WaT + boff), *(const v8bf*)(WaT + boff + 16));
      accA[ni] = wmma_bf16(afrag, bA, accA[ni]);
      v16bf bB = cat8(*(const v8bf*)(WbT + boff), *(const v8bf*)(WbT + boff + 16));
      accB[ni] = wmma_bf16(afrag, bB, accB[ni]);
    }
  }

  // fused epilogue: tanh * sigmoid * Wc, reduce over 256 columns
  float rowsum[8];
#pragma unroll
  for (int e = 0; e < 8; ++e) rowsum[e] = 0.f;
#pragma unroll
  for (int ni = 0; ni < 8; ++ni) {
    int c = c0 + ni * 16 + l16;
    float bav = ba[c], bbv = bb[c], wcv = Wc[c];
#pragma unroll
    for (int e = 0; e < 8; ++e) {
      float av = tanhf(accA[ni][e] + bav);
      float sv = 1.f / (1.f + expf(-(accB[ni][e] + bbv)));
      rowsum[e] += av * sv * wcv;
    }
  }
#pragma unroll
  for (int e = 0; e < 8; ++e)
    for (int off = 8; off >= 1; off >>= 1)
      rowsum[e] += __shfl_xor(rowsum[e], off, 16);   // reduce within 16-lane half

  __shared__ float red[2][64];
  if (l16 == 0) {
#pragma unroll
    for (int e = 0; e < 8; ++e)
      red[wave >> 2][(wave & 3) * 16 + half * 8 + e] = rowsum[e];
  }
  __syncthreads();
  int t = threadIdx.x;
  if (t < 64) {
    int rr = blockIdx.x * 64 + t;
    if (rr < NN) Anb[rr] = red[0][t] + red[1][t];
  }
}

// ---------------- kernel 3: softmax over N + top8/bottom8 indices ------------
__global__ __launch_bounds__(1024) void clam_softmax_topk(const float* __restrict__ Anb,
                                                          const float* __restrict__ bc,
                                                          float* __restrict__ Asm,
                                                          float* __restrict__ Araw_out,
                                                          int* __restrict__ idx) {
  __shared__ float sred[1024];
  __shared__ int   sidx[1024];
  __shared__ int   chosen[16];
  const int t = threadIdx.x;
  const float bcv = bc[0];

  // max
  float m = -3.4e38f;
  for (int i = t; i < NN; i += 1024) m = fmaxf(m, Anb[i] + bcv);
  sred[t] = m; __syncthreads();
  for (int s = 512; s > 0; s >>= 1) { if (t < s) sred[t] = fmaxf(sred[t], sred[t + s]); __syncthreads(); }
  m = sred[0]; __syncthreads();

  // sum + write raw scores to d_out A_raw region
  float z = 0.f;
  for (int i = t; i < NN; i += 1024) {
    float r = Anb[i] + bcv;
    Araw_out[i] = r;
    z += expf(r - m);
  }
  sred[t] = z; __syncthreads();
  for (int s = 512; s > 0; s >>= 1) { if (t < s) sred[t] += sred[t + s]; __syncthreads(); }
  float invZ = 1.f / sred[0]; __syncthreads();
  for (int i = t; i < NN; i += 1024) Asm[i] = expf(Anb[i] + bcv - m) * invZ;

  // iterative top-8 then bottom-8 (ties -> lowest index, like lax.top_k)
  for (int k = 0; k < 16; ++k) {
    const bool topmode = (k < 8);
    const int  base = topmode ? 0 : 8;
    float best = topmode ? -3.4e38f : 3.4e38f;
    int bi = -1;
    for (int i = t; i < NN; i += 1024) {
      bool ex = false;
      for (int j = base; j < k; ++j) if (chosen[j] == i) ex = true;
      if (ex) continue;
      float v = Anb[i];
      bool better = topmode ? (v > best) : (v < best);
      if (better || (v == best && (unsigned)i < (unsigned)bi)) { best = v; bi = i; }
    }
    sred[t] = best; sidx[t] = bi; __syncthreads();
    for (int s = 512; s > 0; s >>= 1) {
      if (t < s) {
        float vo = sred[t + s]; int io = sidx[t + s];
        bool better = topmode ? (vo > sred[t]) : (vo < sred[t]);
        if (better || (vo == sred[t] && (unsigned)io < (unsigned)sidx[t])) { sred[t] = vo; sidx[t] = io; }
      }
      __syncthreads();
    }
    if (t == 0) chosen[k] = sidx[0];
    __syncthreads();
  }
  if (t < 16) idx[t] = chosen[t];
}

// ---------------- kernel 4: Ms[c] = sum_n Asm[n] * x[n][c] -------------------
__global__ __launch_bounds__(256) void clam_colsum(const float* __restrict__ Asm,
                                                   const bf16_t* __restrict__ x,
                                                   float* __restrict__ Ms) {
  __shared__ float sred[256];
  const int c = blockIdx.x;          // 0..511
  float s = 0.f;
  for (int i = threadIdx.x; i < NN; i += 256)
    s += Asm[i] * (float)x[(size_t)i * HD + c];
  sred[threadIdx.x] = s; __syncthreads();
  for (int st = 128; st > 0; st >>= 1) {
    if (threadIdx.x < st) sred[threadIdx.x] += sred[threadIdx.x + st];
    __syncthreads();
  }
  if (threadIdx.x == 0) Ms[c] = sred[0];
}

// ---------------- kernel 5: instance head + NLL loss (16 rows) ---------------
__global__ __launch_bounds__(512) void clam_inst(const bf16_t* __restrict__ x,
                                                 const int* __restrict__ idx,
                                                 const float* __restrict__ Wic,
                                                 const float* __restrict__ bic,
                                                 float* __restrict__ out_loss) {
  __shared__ float lg[16][2];
  const int t = threadIdx.x;
  const int row = t >> 5, lane = t & 31;
  const int n = idx[row];
  float s0 = 0.f, s1 = 0.f;
  for (int k = lane; k < HD; k += 32) {
    float xv = (float)x[(size_t)n * HD + k];
    s0 += xv * Wic[k * 2 + 0];
    s1 += xv * Wic[k * 2 + 1];
  }
  for (int off = 16; off >= 1; off >>= 1) {
    s0 += __shfl_xor(s0, off, 32);
    s1 += __shfl_xor(s1, off, 32);
  }
  if (lane == 0) { lg[row][0] = s0 + bic[0]; lg[row][1] = s1 + bic[1]; }
  __syncthreads();
  if (t == 0) {
    float loss = 0.f;
    for (int r = 0; r < 16; ++r) {
      float l0 = lg[r][0], l1 = lg[r][1];
      float mx = fmaxf(l0, l1);
      float lse = mx + logf(expf(l0 - mx) + expf(l1 - mx));
      float lp = ((r < KS) ? l1 : l0) - lse;    // targets = [1]*8 + [0]*8
      loss -= lp;
    }
    *out_loss = loss / 16.f;
  }
}

// ---------------- kernel 6: gated fusion + classifier ------------------------
__global__ __launch_bounds__(256) void clam_final(const float* __restrict__ Ms,      // [4][512]
                                                  const float* __restrict__ Wattn,   // [2048][4]
                                                  const float* __restrict__ battn,
                                                  const float* __restrict__ Wcls,    // [512][2]
                                                  const float* __restrict__ bcls,
                                                  float* __restrict__ out) {
  __shared__ float s4[4][256];
  __shared__ float wsh[4];
  const int t = threadIdx.x;
  float part[4] = {0.f, 0.f, 0.f, 0.f};
  for (int i = t; i < 4 * HD; i += 256) {
    float v = Ms[i];
#pragma unroll
    for (int j = 0; j < 4; ++j) part[j] += v * Wattn[i * 4 + j];
  }
#pragma unroll
  for (int j = 0; j < 4; ++j) s4[j][t] = part[j];
  __syncthreads();
  for (int s = 128; s > 0; s >>= 1) {
    if (t < s) {
#pragma unroll
      for (int j = 0; j < 4; ++j) s4[j][t] += s4[j][t + s];
    }
    __syncthreads();
  }
  if (t == 0) {
    float sc[4]; float mx = -3.4e38f;
    for (int j = 0; j < 4; ++j) { sc[j] = s4[j][0] + battn[j]; mx = fmaxf(mx, sc[j]); }
    float z = 0.f;
    for (int j = 0; j < 4; ++j) { sc[j] = expf(sc[j] - mx); z += sc[j]; }
    for (int j = 0; j < 4; ++j) wsh[j] = sc[j] / z;
  }
  __syncthreads();
  float p0 = 0.f, p1 = 0.f;
  for (int c = t; c < HD; c += 256) {
    float Mv = wsh[0] * Ms[c] + wsh[1] * Ms[HD + c] + wsh[2] * Ms[2 * HD + c] + wsh[3] * Ms[3 * HD + c];
    p0 += Mv * Wcls[c * 2 + 0];
    p1 += Mv * Wcls[c * 2 + 1];
  }
  s4[0][t] = p0; s4[1][t] = p1; __syncthreads();
  for (int s = 128; s > 0; s >>= 1) {
    if (t < s) { s4[0][t] += s4[0][t + s]; s4[1][t] += s4[1][t + s]; }
    __syncthreads();
  }
  if (t == 0) {
    float l0 = s4[0][0] + bcls[0], l1 = s4[1][0] + bcls[1];
    out[0] = l0; out[1] = l1;
    float mx = fmaxf(l0, l1);
    float e0 = expf(l0 - mx), e1 = expf(l1 - mx), z = e0 + e1;
    out[2] = e0 / z; out[3] = e1 / z;
    out[4] = (l1 > l0) ? 1.0f : 0.0f;   // argmax (first index on tie)
  }
}

// ---------------- host-side launch -------------------------------------------
static inline size_t alignup(size_t v) { return (v + 255) & ~(size_t)255; }

extern "C" void kernel_launch(void* const* d_in, const int* in_sizes, int n_in,
                              void* d_out, int out_size, void* d_ws, size_t ws_size,
                              hipStream_t stream) {
  const float* h[4] = {(const float*)d_in[0], (const float*)d_in[1],
                       (const float*)d_in[2], (const float*)d_in[3]};
  const float* Wfc   = (const float*)d_in[4];
  const float* bfc   = (const float*)d_in[5];
  const float* Wa    = (const float*)d_in[6];
  const float* ba    = (const float*)d_in[7];
  const float* Wb    = (const float*)d_in[8];
  const float* bb    = (const float*)d_in[9];
  const float* Wc    = (const float*)d_in[10];
  const float* bc    = (const float*)d_in[11];
  const float* Wattn = (const float*)d_in[12];
  const float* battn = (const float*)d_in[13];
  const float* Wcls  = (const float*)d_in[14];
  const float* bcls  = (const float*)d_in[15];
  const float* Wic   = (const float*)d_in[16];
  const float* bic   = (const float*)d_in[17];

  // workspace carve (~53 MB)
  char* w = (char*)d_ws;
  size_t off = 0;
  bf16_t* WfcT = (bf16_t*)(w + off); off = alignup(off + (size_t)LD * HD * 2);
  bf16_t* WaT  = (bf16_t*)(w + off); off = alignup(off + (size_t)HD * DD * 2);
  bf16_t* WbT  = (bf16_t*)(w + off); off = alignup(off + (size_t)HD * DD * 2);
  bf16_t* xbuf = (bf16_t*)(w + off); off = alignup(off + (size_t)NN * HD * 2);
  float*  Anb  = (float*)(w + off);  off = alignup(off + (size_t)NN * 4);
  float*  Asm  = (float*)(w + off);  off = alignup(off + (size_t)NN * 4);
  float*  Ms   = (float*)(w + off);  off = alignup(off + (size_t)4 * HD * 4);
  int*    idx  = (int*)(w + off);    off = alignup(off + 64);

  float* out = (float*)d_out;
  // d_out layout (flat, return order): logits[2], Y_prob[2], Y_hat[1], A_raw[50000], inst_loss[1]
  float* out_Araw = out + 5;
  float* out_loss = out + 5 + NN;

  clam_prep<<<(LD * HD + 255) / 256, 256, 0, stream>>>(Wfc, Wa, Wb, WfcT, WaT, WbT);

  for (int f = 0; f < 4; ++f) {
    clam_gemm1<<<391 * 8, 256, 0, stream>>>(h[f], WfcT, bfc, xbuf);
    clam_gemm2<<<(NN + 63) / 64, 256, 0, stream>>>(xbuf, WaT, WbT, ba, bb, Wc, Anb);
    clam_softmax_topk<<<1, 1024, 0, stream>>>(Anb, bc, Asm, out_Araw, idx);
    clam_colsum<<<HD, 256, 0, stream>>>(Asm, xbuf, Ms + f * HD);
    clam_inst<<<1, 512, 0, stream>>>(xbuf, idx, Wic, bic, out_loss);
  }
  clam_final<<<1, 256, 0, stream>>>(Ms, Wattn, battn, Wcls, bcls, out);
}
